// RvNN_75874892251840
// MI455X (gfx1250) — matrix operand order repaired and checked by reading, not compiled, not run on hardware
//
#include <hip/hip_runtime.h>

#define VOCAB 30000
#define HID   512
#define H3    1536
#define NCLS  4
#define NEDGE 4096
#define KW    32
#define NNODE (NEDGE + 1)
#define MT    96   // 1536/16 row tiles

typedef __attribute__((ext_vector_type(16))) __bf16 v16bf;
typedef __attribute__((ext_vector_type(8)))  __bf16 v8bf;
typedef __attribute__((ext_vector_type(8)))  float  v8f;

__device__ __forceinline__ unsigned short f2bf(float f) {
    unsigned int u = __float_as_uint(f);
    u += 0x7FFFu + ((u >> 16) & 1u);   // round-to-nearest-even truncation
    return (unsigned short)(u >> 16);
}

// A-operand fragment: lane = (h, row); K chunks [8h..8h+7] and [16+8h..16+8h+7]
__device__ __forceinline__ v16bf ldA(const unsigned short* p) {
    v8bf lo = *(const v8bf*)p;
    v8bf hi = *(const v8bf*)(p + 16);
    return __builtin_shufflevector(lo, hi, 0,1,2,3,4,5,6,7,8,9,10,11,12,13,14,15);
}

#define WMMA_BF16(A, B, C) \
    __builtin_amdgcn_wmma_f32_16x16x32_bf16(false, (A), false, (B), (short)0, (C), false, false)

// ---------------- f32 -> bf16 convert ----------------
__global__ void k_cvt_bf16(const float* __restrict__ src, unsigned short* __restrict__ dst, int n) {
    int i = blockIdx.x * blockDim.x + threadIdx.x;
    if (i < n) dst[i] = f2bf(src[i]);
}

// ---------------- embedding mix: Xt[e][d] = sum_k word[e,k] * E[d, idx[e,k]] ----------------
__global__ void k_embed(const float* __restrict__ word, const int* __restrict__ word_idx,
                        const float* __restrict__ E, unsigned short* __restrict__ Xt) {
    __shared__ float wsh[KW];
    __shared__ int   ish[KW];
    int e = blockIdx.x;
    int t = threadIdx.x;
    if (t < KW) { wsh[t] = word[e * KW + t]; ish[t] = word_idx[e * KW + t]; }
    __syncthreads();
    for (int d = t; d < HID; d += 128) {
        float acc = 0.f;
        #pragma unroll 8
        for (int k = 0; k < KW; k++)
            acc = fmaf(wsh[k], E[(size_t)d * VOCAB + ish[k]], acc);
        Xt[(size_t)e * HID + d] = f2bf(acc);
    }
}

// ---------------- big GEMM: GIt[n][m] = (W_ih @ X)^T, M=1536 K=512 N=4096 ----------------
// one wave per block, 2(M) x 4(N) register-blocked 16x16 tiles -> 8 WMMA per k-step
__global__ void __launch_bounds__(32)
k_gemm_gi(const unsigned short* __restrict__ Wb,
          const unsigned short* __restrict__ Xt,
          float* __restrict__ GIt) {
    int lane = threadIdx.x;
    int lr = lane & 15;
    int h  = lane >> 4;          // lane half
    int mt0 = blockIdx.x * 2;    // 48 groups
    int nt0 = blockIdx.y * 4;    // 64 groups
    const unsigned short* a0 = Wb + (size_t)(mt0 * 16 + lr) * HID + 8 * h;
    const unsigned short* a1 = a0 + 16 * HID;
    const unsigned short* b0 = Xt + (size_t)(nt0 * 16 + lr) * HID + 8 * h;
    const unsigned short* b1 = b0 + 16 * HID;
    const unsigned short* b2 = b0 + 32 * HID;
    const unsigned short* b3 = b0 + 48 * HID;
    v8f c00 = {}, c01 = {}, c02 = {}, c03 = {};
    v8f c10 = {}, c11 = {}, c12 = {}, c13 = {};
    #pragma unroll 2
    for (int k0 = 0; k0 < HID; k0 += 32) {
        __builtin_prefetch(b0 + k0 + 64, 0, 0);
        __builtin_prefetch(b2 + k0 + 64, 0, 0);
        v16bf A0 = ldA(a0 + k0);
        v16bf A1 = ldA(a1 + k0);
        v16bf B0 = ldA(b0 + k0);
        v16bf B1 = ldA(b1 + k0);
        v16bf B2 = ldA(b2 + k0);
        v16bf B3 = ldA(b3 + k0);
        c00 = WMMA_BF16(A0, B0, c00);
        c01 = WMMA_BF16(A0, B1, c01);
        c02 = WMMA_BF16(A0, B2, c02);
        c03 = WMMA_BF16(A0, B3, c03);
        c10 = WMMA_BF16(A1, B0, c10);
        c11 = WMMA_BF16(A1, B1, c11);
        c12 = WMMA_BF16(A1, B2, c12);
        c13 = WMMA_BF16(A1, B3, c13);
    }
    // D tile: lane holds column (lane&15), rows m0+8h .. m0+8h+7 contiguous
    int m0 = mt0 * 16 + (h << 3);
    int m1 = m0 + 16;
    float* d0 = GIt + (size_t)(nt0 * 16 + lr) * H3;
    *(v8f*)(d0 + m0) = c00;               *(v8f*)(d0 + m1) = c10;
    *(v8f*)(d0 + 16 * H3 + m0) = c01;     *(v8f*)(d0 + 16 * H3 + m1) = c11;
    *(v8f*)(d0 + 32 * H3 + m0) = c02;     *(v8f*)(d0 + 32 * H3 + m1) = c12;
    *(v8f*)(d0 + 48 * H3 + m0) = c03;     *(v8f*)(d0 + 48 * H3 + m1) = c13;
}

// ---------------- level scheduler (single thread, cheap sequential pass) ----------------
__global__ void k_sched(const int* __restrict__ tree, int* depth, int* counts, int* offs,
                        int* level_edges, int* is_parent, int* meta,
                        float* hf, unsigned short* hb) {
    if (threadIdx.x != 0 || blockIdx.x != 0) return;
    for (int i = 0; i < NNODE; i++) is_parent[i] = 0;
    for (int i = 0; i < NNODE + 2; i++) counts[i] = 0;
    depth[0] = 0;
    int maxd = 0;
    for (int e = 0; e < NEDGE; e++) {
        int p = tree[2 * e];
        int d = depth[p] + 1;
        depth[e + 1] = d;
        counts[d]++;
        if (d > maxd) maxd = d;
        is_parent[p] = 1;
    }
    int run = 0;
    for (int L = 0; L <= maxd + 1; L++) { offs[L] = run; run += counts[L]; counts[L] = 0; }
    for (int e = 0; e < NEDGE; e++) {
        int d = depth[e + 1];
        level_edges[offs[d] + counts[d]++] = e;
    }
    meta[0] = maxd;
    for (int d = 0; d < HID; d++) { hf[d] = 0.f; hb[d] = 0; }   // h[root] = 0
}

// ---------------- persistent per-level: GH = W_hh @ h_parent (WMMA, 2x2 tiles) + fused GRU gates ----------------
__global__ void __launch_bounds__(1024)
k_levels(const int* __restrict__ tree, const unsigned short* __restrict__ Whh,
         const float* __restrict__ GIt, const float* __restrict__ b_ih, const float* __restrict__ b_hh,
         float* __restrict__ GHt, float* __restrict__ hf, unsigned short* __restrict__ hb,
         const int* __restrict__ offs, const int* __restrict__ level_edges,
         const int* __restrict__ meta) {
    int tid  = threadIdx.x;
    int lane = tid & 31;
    int wave = tid >> 5;
    int lr = lane & 15;
    int h  = lane >> 4;
    int nlev = meta[0];
    const int mG = MT / 2;   // 48 M-tile pairs

    for (int L = 1; L <= nlev; L++) {
        int base = offs[L];
        int B    = offs[L + 1] - offs[L];
        int nT   = (B + 15) >> 4;
        int nG   = (nT + 1) >> 1;

        for (int t = wave; t < mG * nG; t += 32) {
            int mg = t % mG, ng = t / mG;
            int mt0 = mg * 2, nt0 = ng * 2;
            int n0 = nt0 * 16 + lr;
            int n1 = n0 + 16;
            int p0 = 0, p1 = 0;           // OOB lanes read h row 0 (always zero)
            if (n0 < B) { int e = level_edges[base + n0]; p0 = tree[2 * e]; }
            if (n1 < B) { int e = level_edges[base + n1]; p1 = tree[2 * e]; }
            const unsigned short* a0 = Whh + (size_t)(mt0 * 16 + lr) * HID + 8 * h;
            const unsigned short* a1 = a0 + 16 * HID;
            const unsigned short* b0 = hb + (size_t)p0 * HID + 8 * h;
            const unsigned short* b1 = hb + (size_t)p1 * HID + 8 * h;
            v8f c00 = {}, c01 = {}, c10 = {}, c11 = {};
            #pragma unroll 2
            for (int k0 = 0; k0 < HID; k0 += 32) {
                v16bf A0 = ldA(a0 + k0);
                v16bf A1 = ldA(a1 + k0);
                v16bf B0 = ldA(b0 + k0);
                v16bf B1 = ldA(b1 + k0);
                c00 = WMMA_BF16(A0, B0, c00);
                c01 = WMMA_BF16(A0, B1, c01);
                c10 = WMMA_BF16(A1, B0, c10);
                c11 = WMMA_BF16(A1, B1, c11);
            }
            int m0 = mt0 * 16 + (h << 3);
            int m1 = m0 + 16;
            if (n0 < B) {
                float* d = GHt + (size_t)n0 * H3;
                *(v8f*)(d + m0) = c00;
                *(v8f*)(d + m1) = c10;
            }
            if (n1 < B) {
                float* d = GHt + (size_t)n1 * H3;
                *(v8f*)(d + m0) = c01;
                *(v8f*)(d + m1) = c11;
            }
        }
        __threadfence();
        __syncthreads();

        // GRU gates for this level's edges
        for (int i = tid; i < B * HID; i += 1024) {
            int b = i >> 9;            // edge slot within level
            int d = i & 511;           // hidden dim
            int e = level_edges[base + b];
            int p = tree[2 * e];
            int c = e + 1;
            float gir = GIt[(size_t)e * H3 + d]           + b_ih[d];
            float giz = GIt[(size_t)e * H3 + HID + d]     + b_ih[HID + d];
            float gin = GIt[(size_t)e * H3 + 2 * HID + d] + b_ih[2 * HID + d];
            float ghr = GHt[(size_t)b * H3 + d]           + b_hh[d];
            float ghz = GHt[(size_t)b * H3 + HID + d]     + b_hh[HID + d];
            float ghn = GHt[(size_t)b * H3 + 2 * HID + d] + b_hh[2 * HID + d];
            float r  = 1.f / (1.f + __expf(-(gir + ghr)));
            float z  = 1.f / (1.f + __expf(-(giz + ghz)));
            float nn = tanhf(gin + r * ghn);
            float hp = hf[(size_t)p * HID + d];
            float hv = (1.f - z) * nn + z * hp;
            hf[(size_t)c * HID + d] = hv;
            hb[(size_t)c * HID + d] = f2bf(hv);
        }
        __threadfence();
        __syncthreads();
    }
}

// ---------------- leaf max, logits, softmax ----------------
__global__ void __launch_bounds__(512)
k_final(const float* __restrict__ hf, const int* __restrict__ is_parent,
        const float* __restrict__ out_W, const float* __restrict__ out_b,
        float* __restrict__ out) {
    __shared__ float fs[HID];
    __shared__ float red[512];
    int t = threadIdx.x;
    float m = -1e30f;
    for (int nd = 0; nd < NNODE; nd++) {
        if (!is_parent[nd]) m = fmaxf(m, hf[(size_t)nd * HID + t]);
    }
    fs[t] = m;
    __syncthreads();
    int c = t >> 7;       // class 0..3
    int j = t & 127;
    float part = 0.f;
    #pragma unroll
    for (int q = 0; q < 4; q++) {
        int d = j + q * 128;
        part = fmaf(out_W[c * HID + d], fs[d], part);
    }
    red[t] = part;
    __syncthreads();
    if (t < NCLS) {
        float s = 0.f;
        for (int k = 0; k < 128; k++) s += red[t * 128 + k];
        red[t] = s + out_b[t];
    }
    __syncthreads();
    if (t == 0) {
        float l0 = red[0], l1 = red[1], l2 = red[2], l3 = red[3];
        float mx = fmaxf(fmaxf(l0, l1), fmaxf(l2, l3));
        float e0 = __expf(l0 - mx), e1 = __expf(l1 - mx), e2 = __expf(l2 - mx), e3 = __expf(l3 - mx);
        float s = e0 + e1 + e2 + e3;
        out[0] = e0 / s; out[1] = e1 / s; out[2] = e2 / s; out[3] = e3 / s;
    }
}

extern "C" void kernel_launch(void* const* d_in, const int* in_sizes, int n_in,
                              void* d_out, int out_size, void* d_ws, size_t ws_size,
                              hipStream_t stream) {
    const int*   tree     = (const int*)d_in[0];
    const float* word     = (const float*)d_in[1];
    const int*   word_idx = (const int*)d_in[2];
    const float* E_tab    = (const float*)d_in[3];
    const float* W_ih     = (const float*)d_in[4];
    const float* W_hh     = (const float*)d_in[5];
    const float* b_ih     = (const float*)d_in[6];
    const float* b_hh     = (const float*)d_in[7];
    const float* out_W    = (const float*)d_in[8];
    const float* out_b    = (const float*)d_in[9];
    float* out = (float*)d_out;

    char* p = (char*)d_ws;
    auto alloc = [&](size_t bytes) -> char* {
        char* r = p; p += (bytes + 255) & ~(size_t)255; return r;
    };
    unsigned short* Wih_b = (unsigned short*)alloc((size_t)H3 * HID * 2);
    unsigned short* Whh_b = (unsigned short*)alloc((size_t)H3 * HID * 2);
    unsigned short* Xt    = (unsigned short*)alloc((size_t)NEDGE * HID * 2);
    float*          GIt   = (float*)alloc((size_t)NEDGE * H3 * 4);
    float*          GHt   = (float*)alloc((size_t)NEDGE * H3 * 4);
    float*          hf    = (float*)alloc((size_t)NNODE * HID * 4);
    unsigned short* hb    = (unsigned short*)alloc((size_t)NNODE * HID * 2);
    int* depth       = (int*)alloc((NNODE + 16) * 4);
    int* counts      = (int*)alloc((NNODE + 16) * 4);
    int* offs        = (int*)alloc((NNODE + 16) * 4);
    int* level_edges = (int*)alloc((NEDGE + 32) * 4);
    int* is_parent   = (int*)alloc((NNODE + 16) * 4);
    int* meta        = (int*)alloc(64);

    k_cvt_bf16<<<(H3 * HID + 255) / 256, 256, 0, stream>>>(W_ih, Wih_b, H3 * HID);
    k_cvt_bf16<<<(H3 * HID + 255) / 256, 256, 0, stream>>>(W_hh, Whh_b, H3 * HID);
    k_embed<<<NEDGE, 128, 0, stream>>>(word, word_idx, E_tab, Xt);
    dim3 g(MT / 2, NEDGE / 16 / 4);   // 48 x 64 waves, each 32x64 output block
    k_gemm_gi<<<g, 32, 0, stream>>>(Wih_b, Xt, GIt);
    k_sched<<<1, 1, 0, stream>>>(tree, depth, counts, offs, level_edges, is_parent, meta, hf, hb);
    k_levels<<<1, 1024, 0, stream>>>(tree, Whh_b, GIt, b_ih, b_hh, GHt, hf, hb,
                                     offs, level_edges, meta);
    k_final<<<1, 512, 0, stream>>>(hf, is_parent, out_W, out_b, out);
}
